// Block_softmoe_20409684591350
// MI455X (gfx1250) — matrix operand order repaired and checked by reading, not compiled, not run on hardware
//
#include <hip/hip_runtime.h>
#include <math.h>

// ---------------------------------------------------------------------------
// CDNA5 / gfx1250 transformer block (Block_softmoe 'a'-path), bf16 WMMA.
// B=4, S=2048, C=256, H=8, hd=32.  Flash-attention style: no SxS intermediate.
// Round 4: pre-convert x and weights to bf16 (kills all inner-loop cvt);
// 32x64 register-blocked GEMM (8 WMMAs per k-step, B frags reused 2x);
// vectorized A/B/P fragment loads (uint4 / b128); async K->LDS staging kept.
// ---------------------------------------------------------------------------

typedef __attribute__((ext_vector_type(16))) __bf16 v16bf;
typedef __attribute__((ext_vector_type(8)))  float  v8f;

#define BB   4
#define SS   2048
#define CC   256
#define HH   8
#define HD   32

#if defined(__AMDGCN__) && __has_builtin(__builtin_amdgcn_global_load_async_to_lds_b64)
#define USE_ASYNC_LDS 1
typedef int v2i_vs __attribute__((vector_size(8)));
typedef __attribute__((address_space(1))) v2i_vs* as1_v2i_p;
typedef __attribute__((address_space(3))) v2i_vs* as3_v2i_p;
#else
#define USE_ASYNC_LDS 0
#endif

static __device__ inline v8f wmma_bf16(v16bf a, v16bf b, v8f c) {
  // D(f32 16x16) = A(bf16 16x32) * B(bf16 32x16) + C
  return __builtin_amdgcn_wmma_f32_16x16x32_bf16(false, a, false, b,
                                                 (short)0, c, false, false);
}

union bf4u { uint2 u; __bf16 b[4]; };
union v16u { v16bf v; uint4 q[2]; };

// 16 bf16 (32B) from a 16B-aligned pointer (LDS rows padded to 80B).
static __device__ inline v16bf load_v16bf_a16(const __bf16* p) {
  v16u r;
  r.q[0] = *(const uint4*)(p);
  r.q[1] = *(const uint4*)(p + 8);
  return r.v;
}

// A-fragment 16x32 bf16 for row `arow`: K {16g..16g+15} is contiguous
// -> exactly two 16B loads.
static __device__ inline v16bf load_afrag(const __bf16* arow, int k0, int g) {
  v16u r;
  r.q[0] = *(const uint4*)(arow + k0 + (g << 4));
  r.q[1] = *(const uint4*)(arow + k0 + 8 + (g << 4));
  return r.v;
}

// ---------------------------------------------------------------------------
// Kernel 0: fp32 -> bf16 elementwise convert (8 elems/thread, n % 8 == 0).
// ---------------------------------------------------------------------------
__global__ __launch_bounds__(256) void cvt_bf16_kernel(
    const float* __restrict__ src, __bf16* __restrict__ dst, int n) {
  int i = ((int)blockIdx.x * 256 + (int)threadIdx.x) * 8;
  if (i >= n) return;
  float4 a = *(const float4*)(src + i);
  float4 b = *(const float4*)(src + i + 4);
  union { __bf16 e[8]; uint4 u; } o;
  o.e[0] = (__bf16)a.x; o.e[1] = (__bf16)a.y;
  o.e[2] = (__bf16)a.z; o.e[3] = (__bf16)a.w;
  o.e[4] = (__bf16)b.x; o.e[5] = (__bf16)b.y;
  o.e[6] = (__bf16)b.z; o.e[7] = (__bf16)b.w;
  *(uint4*)(dst + i) = o.u;
}

// ---------------------------------------------------------------------------
// Register-blocked wave GEMM: 32(M) x 64(N) at (m0,n0),
// A[m][k] bf16 (lda) x W[n][k]^T bf16 (ldw).  Pure-bf16 operand stream:
// per k-step 4 A loads + 8 B loads (b128) feed 8 WMMAs, zero VALU converts.
// ---------------------------------------------------------------------------
static __device__ inline void gemm32x64_bf16(const __bf16* __restrict__ A, int lda,
                                             const __bf16* __restrict__ W, int ldw,
                                             int m0, int n0, int Kdim, int lane,
                                             v8f acc[8]) {
  const int g = lane >> 4, lr = lane & 15;
  const __bf16* arow0 = A + (size_t)(m0 + lr) * lda;
  const __bf16* arow1 = A + (size_t)(m0 + 16 + lr) * lda;
  const __bf16* wrow  = W + (size_t)(n0 + lr) * ldw + (g << 4);
  for (int k0 = 0; k0 < Kdim; k0 += 32) {
    v16bf af0 = load_afrag(arow0, k0, g);
    v16bf af1 = load_afrag(arow1, k0, g);
#pragma unroll
    for (int t = 0; t < 4; ++t) {
      v16bf bf = *(const v16bf*)(wrow + (size_t)(t << 4) * ldw + k0);  // 32B aligned
      acc[t]     = wmma_bf16(af0, bf, acc[t]);
      acc[4 + t] = wmma_bf16(af1, bf, acc[4 + t]);
    }
  }
}

// ---------------------------------------------------------------------------
// Kernel 1: fused QKV projection from bf16 x.  q scaled by hd^-0.5,
// stored [B,H,S,hd] bf16.  256 m-tiles x 12 n-groups = 3072 waves -> 384 blk.
// ---------------------------------------------------------------------------
__global__ __launch_bounds__(256) void qkv_kernel(
    const __bf16* __restrict__ xb,
    const __bf16* __restrict__ Wq, const float* __restrict__ bq,
    const __bf16* __restrict__ Wk, const float* __restrict__ bk,
    const __bf16* __restrict__ Wv, const float* __restrict__ bv,
    __bf16* __restrict__ Qo, __bf16* __restrict__ Ko, __bf16* __restrict__ Vo) {
  const int wave = (int)((blockIdx.x * blockDim.x + threadIdx.x) >> 5);
  const int lane = (int)(threadIdx.x & 31);
  const int mt = wave / 12;
  const int ngrp = wave % 12;
  const int which = ngrp >> 2;          // 0=Q 1=K 2=V
  const int n0 = (ngrp & 3) << 6;
  const int m0 = mt << 5;
  const __bf16* W    = (which == 0) ? Wq : (which == 1) ? Wk : Wv;
  const float*  bias = (which == 0) ? bq : (which == 1) ? bk : bv;
  __bf16*       out  = (which == 0) ? Qo : (which == 1) ? Ko : Vo;

  v8f acc[8] = {};
  gemm32x64_bf16(xb, CC, W, CC, m0, n0, CC, lane, acc);

  const int g = lane >> 4, lr = lane & 15;
  const float sc = (which == 0) ? 0.17677669529663687f : 1.0f;  // hd^-0.5
#pragma unroll
  for (int t = 0; t < 4; ++t) {
    const int col = n0 + (t << 4) + lr;     // output channel 0..255
    const float bv_ = bias[col];
    const int h = col >> 5, d = col & 31;
#pragma unroll
    for (int sub = 0; sub < 2; ++sub) {
#pragma unroll
      for (int r = 0; r < 8; ++r) {         // D layout: row = r + 8*g
        int m = m0 + (sub << 4) + r + (g << 3);
        int b = m >> 11, s = m & (SS - 1);
        float v = (acc[(sub << 2) + t][r] + bv_) * sc;
        out[(((size_t)b * HH + h) * SS + s) * HD + d] = (__bf16)v;
      }
    }
  }
}

// ---------------------------------------------------------------------------
// Kernel 2: flash attention.  8 waves/block share one (b,h); each wave owns a
// 16-query tile.  Per 32-key chunk the block cooperatively stages:
//   Ksh [32 key][32 hd]  (async global->LDS)
//   Vsh [32 hd][32 key]  (transposed -> contiguous P*V B-fragments)
// Rows padded to 40 bf16 (80B) for conflict-free b128 LDS reads.
// Emits fp32 x_out (residual) and bf16 x_out (mlp1 A operand).
// ---------------------------------------------------------------------------
__global__ __launch_bounds__(256) void attn_kernel(
    const __bf16* __restrict__ Q, const __bf16* __restrict__ K,
    const __bf16* __restrict__ V, const int* __restrict__ mask,
    float* __restrict__ xout, __bf16* __restrict__ xob) {
  __shared__ __bf16 Ksh[32 * 40];
  __shared__ __bf16 Vsh[32 * 40];
  __shared__ __bf16 psh[8][16 * 40];   // per-wave P tile (D->A relayout)

  const int tid  = (int)threadIdx.x;
  const int wid  = tid >> 5;
  const int lane = tid & 31;
  const int bh = (int)blockIdx.x >> 4;              // 32 (b,h) pairs
  const int qt = (((int)blockIdx.x & 15) << 3) + wid;
  const int b = bh >> 3, h = bh & 7;
  const int q0 = qt << 4;
  const __bf16* Qb = Q + (size_t)bh * SS * HD;
  const __bf16* Kb = K + (size_t)bh * SS * HD;
  const __bf16* Vb = V + (size_t)bh * SS * HD;
  const int* mb = mask + (size_t)b * 3 * SS;        // modality 'a' slice
  const int g = lane >> 4, lr = lane & 15;

  // staging roles: 256 threads move 32x32 bf16 (2KB) per matrix per chunk
  const int skey  = tid >> 3;          // key row 0..31
  const int spart = tid & 7;           // 8B segment within the 64B key row
  const int vhd4  = (tid & 7) << 2;    // 4 consecutive hd elements

  // Q fragment (A layout, K-dim == hd == 32): two 16B loads.
  v16bf qf = load_afrag(Qb + (size_t)(q0 + lr) * HD, 0, g);

  float mrun[8], lrun[8];
  v8f acc0 = {}, acc1 = {};
#pragma unroll
  for (int r = 0; r < 8; ++r) { mrun[r] = -3.0e38f; lrun[r] = 0.0f; }

  __bf16* ps = &psh[wid][0];

  for (int kb = 0; kb < SS; kb += 32) {
    // ---- cooperative staging --------------------------------------------
    {
      const __bf16* gk = Kb + (size_t)(kb + skey) * HD + (spart << 2);
      __bf16* lk = &Ksh[skey * 40 + (spart << 2)];
#if USE_ASYNC_LDS
      __builtin_amdgcn_global_load_async_to_lds_b64(
          (as1_v2i_p)gk, (as3_v2i_p)lk, 0, 0);
#else
      *(uint2*)lk = *(const uint2*)gk;
#endif
      bf4u vv;
      vv.u = *(const uint2*)(Vb + (size_t)(kb + skey) * HD + vhd4);
#pragma unroll
      for (int j = 0; j < 4; ++j)
        Vsh[(vhd4 + j) * 40 + skey] = vv.b[j];   // transpose into [hd][key]
    }
#if USE_ASYNC_LDS
#if __has_builtin(__builtin_amdgcn_s_wait_asynccnt)
    __builtin_amdgcn_s_wait_asynccnt(0);
#else
    asm volatile("s_wait_asynccnt 0" ::: "memory");
#endif
#endif
    __syncthreads();

    // ---- scores: Q[16x32] * K^T[32x16] for both key halves ---------------
    v8f s0 = {}, s1 = {};
    s0 = wmma_bf16(qf, load_v16bf_a16(&Ksh[lr * 40 + (g << 4)]), s0);
    s1 = wmma_bf16(qf, load_v16bf_a16(&Ksh[(16 + lr) * 40 + (g << 4)]), s1);

    const int msk0 = mb[kb + lr];
    const int msk1 = mb[kb + 16 + lr];
    float p0[8], p1[8];
#pragma unroll
    for (int r = 0; r < 8; ++r) {
      float a0 = msk0 ? s0[r] : -3.0e38f;
      float a1 = msk1 ? s1[r] : -3.0e38f;
      float t = fmaxf(a0, a1);
#pragma unroll
      for (int off = 1; off < 16; off <<= 1)   // row max across 16 key lanes
        t = fmaxf(t, __shfl_xor(t, off, 32));
      float mnew = fmaxf(mrun[r], t);
      float corr = __expf(mrun[r] - mnew);
      float e0 = (a0 < -1.0e37f) ? 0.0f : __expf(a0 - mnew);
      float e1 = (a1 < -1.0e37f) ? 0.0f : __expf(a1 - mnew);
      float rs = e0 + e1;
#pragma unroll
      for (int off = 1; off < 16; off <<= 1)   // row sum
        rs += __shfl_xor(rs, off, 32);
      lrun[r] = lrun[r] * corr + rs;
      mrun[r] = mnew;
      acc0[r] *= corr;
      acc1[r] *= corr;
      p0[r] = e0;
      p1[r] = e1;
    }

    // ---- D-layout P -> LDS -> A-layout fragment (intra-wave DS in-order) --
#pragma unroll
    for (int r = 0; r < 8; ++r) {
      int row = r + (g << 3);
      ps[row * 40 + lr]      = (__bf16)p0[r];
      ps[row * 40 + 16 + lr] = (__bf16)p1[r];
    }
    v16bf pa = load_afrag(ps + lr * 40, 0, g);   // two ds_load_b128

    // ---- P[16x32] * V[32x16]: contiguous B-fragments from transposed Vsh --
    v16bf vf0 = load_v16bf_a16(&Vsh[lr * 40 + (g << 4)]);
    v16bf vf1 = load_v16bf_a16(&Vsh[(16 + lr) * 40 + (g << 4)]);
    acc0 = wmma_bf16(pa, vf0, acc0);
    acc1 = wmma_bf16(pa, vf1, acc1);

    __syncthreads();   // protect Ksh/Vsh before next chunk overwrites
  }

  // Epilogue: normalize (NaN->0 via l==0 guard), scatter fp32 + bf16 copies.
#pragma unroll
  for (int r = 0; r < 8; ++r) {
    int q = q0 + r + (g << 3);
    float inv = (lrun[r] > 0.0f) ? (1.0f / lrun[r]) : 0.0f;
    size_t base = ((size_t)b * SS + q) * CC + h * HD;
    float o0 = acc0[r] * inv;
    float o1 = acc1[r] * inv;
    xout[base + lr]      = o0;
    xout[base + 16 + lr] = o1;
    xob[base + lr]       = (__bf16)o0;
    xob[base + 16 + lr]  = (__bf16)o1;
  }
}

// ---------------------------------------------------------------------------
// Kernel 3: h = gelu_exact(x_out @ W1^T + b1), bf16 out.  32x64 wave tiles.
// ---------------------------------------------------------------------------
__global__ __launch_bounds__(256) void mlp1_kernel(
    const __bf16* __restrict__ xob, const __bf16* __restrict__ W1,
    const float* __restrict__ b1, __bf16* __restrict__ hbuf) {
  const int wave = (int)((blockIdx.x * blockDim.x + threadIdx.x) >> 5);
  const int lane = (int)(threadIdx.x & 31);
  const int m0 = (wave >> 2) << 5;
  const int n0 = (wave & 3) << 6;
  v8f acc[8] = {};
  gemm32x64_bf16(xob, CC, W1, CC, m0, n0, CC, lane, acc);
  const int g = lane >> 4, lr = lane & 15;
#pragma unroll
  for (int t = 0; t < 4; ++t) {
    const int col = n0 + (t << 4) + lr;
    const float bias = b1[col];
#pragma unroll
    for (int sub = 0; sub < 2; ++sub) {
#pragma unroll
      for (int r = 0; r < 8; ++r) {
        float v = acc[(sub << 2) + t][r] + bias;
        float ge = 0.5f * v * (1.0f + erff(v * 0.70710678118654752f));
        hbuf[(size_t)(m0 + (sub << 4) + r + (g << 3)) * CC + col] = (__bf16)ge;
      }
    }
  }
}

// ---------------------------------------------------------------------------
// Kernel 4: out = x_out + (h @ W2^T + b2).  32x64 wave tiles.
// ---------------------------------------------------------------------------
__global__ __launch_bounds__(256) void mlp2_kernel(
    const float* __restrict__ xo, const __bf16* __restrict__ hbuf,
    const __bf16* __restrict__ W2, const float* __restrict__ b2,
    float* __restrict__ out) {
  const int wave = (int)((blockIdx.x * blockDim.x + threadIdx.x) >> 5);
  const int lane = (int)(threadIdx.x & 31);
  const int m0 = (wave >> 2) << 5;
  const int n0 = (wave & 3) << 6;
  v8f acc[8] = {};
  gemm32x64_bf16(hbuf, CC, W2, CC, m0, n0, CC, lane, acc);
  const int g = lane >> 4, lr = lane & 15;
#pragma unroll
  for (int t = 0; t < 4; ++t) {
    const int col = n0 + (t << 4) + lr;
    const float bias = b2[col];
#pragma unroll
    for (int sub = 0; sub < 2; ++sub) {
#pragma unroll
      for (int r = 0; r < 8; ++r) {
        size_t idx = (size_t)(m0 + (sub << 4) + r + (g << 3)) * CC + col;
        out[idx] = xo[idx] + acc[(sub << 2) + t][r] + bias;
      }
    }
  }
}

// ---------------------------------------------------------------------------
extern "C" void kernel_launch(void* const* d_in, const int* in_sizes, int n_in,
                              void* d_out, int out_size, void* d_ws, size_t ws_size,
                              hipStream_t stream) {
  const float* x  = (const float*)d_in[0];
  const int* mask = (const int*)d_in[1];
  const float* Wq = (const float*)d_in[2];
  const float* bq = (const float*)d_in[3];
  const float* Wk = (const float*)d_in[4];
  const float* bk = (const float*)d_in[5];
  const float* Wv = (const float*)d_in[6];
  const float* bv = (const float*)d_in[7];
  const float* W1 = (const float*)d_in[8];
  const float* b1 = (const float*)d_in[9];
  const float* W2 = (const float*)d_in[10];
  const float* b2 = (const float*)d_in[11];

  const size_t nElem = (size_t)BB * SS * CC;   // 2,097,152
  const size_t nW    = (size_t)CC * CC;        // 65,536
  char* ws = (char*)d_ws;
  __bf16* Qb  = (__bf16*)(ws);                   // 4 MB
  __bf16* Kb  = (__bf16*)(ws + 2 * nElem);       // 4 MB
  __bf16* Vb  = (__bf16*)(ws + 4 * nElem);       // 4 MB
  float*  xo  = (float*)(ws + 6 * nElem);        // 8 MB (fp32 x_out, residual)
  __bf16* xob = (__bf16*)(ws + 10 * nElem);      // 4 MB (bf16 x_out, mlp1 A)
  __bf16* hb  = (__bf16*)(ws + 12 * nElem);      // 4 MB (bf16 h, mlp2 A)
  __bf16* xbb = (__bf16*)(ws + 14 * nElem);      // 4 MB (bf16 x, qkv A)
  __bf16* wqb = (__bf16*)(ws + 16 * nElem);
  __bf16* wkb = wqb + nW;
  __bf16* wvb = wkb + nW;
  __bf16* w1b = wvb + nW;
  __bf16* w2b = w1b + nW;

  // one-time bf16 conversions (x: 1024 blocks; each weight: 32 blocks)
  cvt_bf16_kernel<<<1024, 256, 0, stream>>>(x,  xbb, (int)nElem);
  cvt_bf16_kernel<<<32,   256, 0, stream>>>(Wq, wqb, (int)nW);
  cvt_bf16_kernel<<<32,   256, 0, stream>>>(Wk, wkb, (int)nW);
  cvt_bf16_kernel<<<32,   256, 0, stream>>>(Wv, wvb, (int)nW);
  cvt_bf16_kernel<<<32,   256, 0, stream>>>(W1, w1b, (int)nW);
  cvt_bf16_kernel<<<32,   256, 0, stream>>>(W2, w2b, (int)nW);

  // 256 m-tiles x 12 n-groups = 3072 waves -> 384 blocks
  qkv_kernel<<<384, 256, 0, stream>>>(xbb, wqb, bq, wkb, bk, wvb, bv, Qb, Kb, Vb);
  // B*H*(S/16) = 4096 waves -> 512 blocks (8 waves/block share one (b,h))
  attn_kernel<<<512, 256, 0, stream>>>(Qb, Kb, Vb, mask, xo, xob);
  // 256 m-tiles x 4 n-groups = 1024 waves -> 128 blocks each
  mlp1_kernel<<<128, 256, 0, stream>>>(xob, w1b, b1, hb);
  mlp2_kernel<<<128, 256, 0, stream>>>(xo, hb, w2b, b2, (float*)d_out);
}
